// KernelRBF_74474732913139
// MI455X (gfx1250) — matrix operand order repaired and checked by reading, not compiled, not run on hardware
//
#include <hip/hip_runtime.h>
#include <hip/hip_bf16.h>

// ---------------------------------------------------------------------------
// RBF kernel: out[i][j] = exp(-GAMMA * (||x1_i||^2 - 2*x1_i.x2_j + ||x2_j||^2))
// x1, x2: [8192, 128] fp32.  out: [8192, 8192] fp32.
//
// MI455X strategy:
//  * Output store (256 MB @ 23.3 TB/s ~= 11 us) is the HBM floor; it exceeds
//    the 192 MB L2 and is write-once -> non-temporal b128 stores (LDS repack
//    for coalescing) so it doesn't evict the operands.
//  * Inputs (8 MB as bf16 hi/lo) live in L2 -> operands loaded from global.
//  * Cross term = 8192x8192x128 GEMM done with V_WMMA_F32_16X16X32_BF16 using a
//    bf16 hi/lo split (hi*hi + hi*lo + lo*hi) for ~fp32 accuracy at bf16-WMMA
//    rate (~52 GFLOP of WMMA work ~= the store time -> balanced).
// ---------------------------------------------------------------------------

#define N_ROWS 8192
#define K_DIM  128
#define GAMMA  0.005f   // 1 / (2 * 10^2)

typedef __attribute__((ext_vector_type(16))) __bf16 v16bf;
typedef __attribute__((ext_vector_type(8)))  float  v8f;
typedef __attribute__((ext_vector_type(4)))  float  f4;

union Frag16 {
    v16bf v;
    uint4 q[2];
};

// ---------------------------------------------------------------------------
// Prep: one wave (32 lanes) per row; lane i owns elements [4i, 4i+4).
// Computes row norm (shfl-xor reduction) and bf16 hi/lo decomposition:
//   hi = bf16(x), lo = bf16(x - float(hi))  =>  hi + lo carries ~16 mantissa bits
// ---------------------------------------------------------------------------
__global__ __launch_bounds__(256) void rbf_prep_kernel(
    const float* __restrict__ x,
    __hip_bfloat16* __restrict__ hi,
    __hip_bfloat16* __restrict__ lo,
    float* __restrict__ norms)
{
    const int lane = threadIdx.x & 31;
    const int wave = threadIdx.x >> 5;
    const int row  = blockIdx.x * 8 + wave;   // 8 waves per block, 1 row each

    const f4* xr = (const f4*)(x + (size_t)row * K_DIM);
    f4 v = xr[lane];

    float s = v[0] * v[0] + v[1] * v[1] + v[2] * v[2] + v[3] * v[3];
#pragma unroll
    for (int off = 16; off > 0; off >>= 1)
        s += __shfl_xor(s, off, 32);
    if (lane == 0) norms[row] = s;

    union { __hip_bfloat16 h[4]; uint2 u; } hp, lp;
#pragma unroll
    for (int j = 0; j < 4; ++j) {
        float f = v[j];
        __hip_bfloat16 h = __float2bfloat16(f);
        hp.h[j] = h;
        lp.h[j] = __float2bfloat16(f - __bfloat162float(h));
    }
    ((uint2*)(hi + (size_t)row * K_DIM))[lane] = hp.u;
    ((uint2*)(lo + (size_t)row * K_DIM))[lane] = lp.u;
}

// ---------------------------------------------------------------------------
// Main GEMM + epilogue.
// Block = 256 threads = 8 waves. Block tile = 128 rows x 64 cols.
// Wave w owns rows [w*16, w*16+16) x 64 cols (4 accumulator tiles).
// K loop: 4 chunks of 32, 3 WMMAs per chunk per tile (hi*hi, hi*lo, lo*hi).
//
// Wave32 operand layouts (CDNA5 ISA 7.12.2):
//  A (16x32 bf16): lane L holds row M=L%16;
//    lanes 0-15:  K in {k0..k0+7} (v0-3) and {k0+16..k0+23} (v4-7)
//    lanes 16-31: K in {k0+8..k0+15}     and {k0+24..k0+31}
//  B (32x16 bf16): lane L holds col N=L%16;
//    lanes 0-15:  K in {k0..k0+7},{k0+16..k0+23}; lanes 16-31: {k0+8..k0+15},{k0+24..k0+31}
//  C/D (16x16 f32): VGPR r, lane L -> M = r + 8*(L/16), N = L%16
//
// Epilogue: apply norms + exp in registers, repack the 128x64 block tile
// through LDS, stream out as non-temporal float4 stores (coalesced).
// ---------------------------------------------------------------------------
__global__ __launch_bounds__(256) void rbf_gemm_kernel(
    const __hip_bfloat16* __restrict__ x1hi,
    const __hip_bfloat16* __restrict__ x1lo,
    const __hip_bfloat16* __restrict__ x2hi,
    const __hip_bfloat16* __restrict__ x2lo,
    const float* __restrict__ n1,
    const float* __restrict__ n2,
    float* __restrict__ out)
{
    __shared__ __align__(16) float tile[128 * 64];

    const int lane   = threadIdx.x & 31;
    const int wave   = threadIdx.x >> 5;
    const int lane16 = lane & 15;
    const bool hiHalf = lane >= 16;

    const int row_base = blockIdx.y * 128 + wave * 16;   // this wave's 16 rows
    const int col_base = blockIdx.x * 64;                // this wave's 64 cols

    v8f acc[4];
#pragma unroll
    for (int t = 0; t < 4; ++t) acc[t] = (v8f)(0.0f);

    // Per-lane base pointers for A fragments (row = row_base + lane16)
    const __hip_bfloat16* a_hi_row = x1hi + (size_t)(row_base + lane16) * K_DIM;
    const __hip_bfloat16* a_lo_row = x1lo + (size_t)(row_base + lane16) * K_DIM;
    const int a_koff = hiHalf ? 8 : 0;    // A: lane half picks K sub-block
    const int b_koff = hiHalf ? 8 : 0;    // B: lane half picks K sub-block (same as A)

#pragma unroll
    for (int k0 = 0; k0 < K_DIM; k0 += 32) {
        // ---- A fragments (shared across the 4 N tiles) ----
        Frag16 Ahi, Alo;
        {
            const __hip_bfloat16* pa = a_hi_row + k0 + a_koff;
            Ahi.q[0] = *(const uint4*)(pa);
            Ahi.q[1] = *(const uint4*)(pa + 16);
            const __hip_bfloat16* pl = a_lo_row + k0 + a_koff;
            Alo.q[0] = *(const uint4*)(pl);
            Alo.q[1] = *(const uint4*)(pl + 16);
        }

#pragma unroll
        for (int t = 0; t < 4; ++t) {
            const int col = col_base + t * 16 + lane16;   // B col for this lane
            const __hip_bfloat16* pbh = x2hi + (size_t)col * K_DIM + k0 + b_koff;
            const __hip_bfloat16* pbl = x2lo + (size_t)col * K_DIM + k0 + b_koff;

            Frag16 Bhi, Blo;
            Bhi.q[0] = *(const uint4*)(pbh);
            Bhi.q[1] = *(const uint4*)(pbh + 16);
            Blo.q[0] = *(const uint4*)(pbl);
            Blo.q[1] = *(const uint4*)(pbl + 16);

            // cross ~= hi*hi + hi*lo + lo*hi   (lo*lo ~ 2^-32, dropped)
            acc[t] = __builtin_amdgcn_wmma_f32_16x16x32_bf16(
                false, Ahi.v, false, Bhi.v, (short)0, acc[t], false, false);
            acc[t] = __builtin_amdgcn_wmma_f32_16x16x32_bf16(
                false, Ahi.v, false, Blo.v, (short)0, acc[t], false, false);
            acc[t] = __builtin_amdgcn_wmma_f32_16x16x32_bf16(
                false, Alo.v, false, Bhi.v, (short)0, acc[t], false, false);
        }
    }

    // ---- Epilogue math in registers: exp(-GAMMA*(n1 - 2*cross + n2)) ----
    float nn1[8];
#pragma unroll
    for (int r = 0; r < 8; ++r) {
        const int row = row_base + r + (hiHalf ? 8 : 0);
        nn1[r] = n1[row];
    }

#pragma unroll
    for (int t = 0; t < 4; ++t) {
        const int col = col_base + t * 16 + lane16;
        const float nn2 = n2[col];
        const int col_l = t * 16 + lane16;
#pragma unroll
        for (int r = 0; r < 8; ++r) {
            const int row_l = wave * 16 + r + (hiHalf ? 8 : 0);
            const float ns = nn1[r] - 2.0f * acc[t][r] + nn2;
            tile[row_l * 64 + col_l] = __expf(-GAMMA * ns);
        }
    }

    __syncthreads();

    // ---- Streamout: 8 coalesced non-temporal float4 stores per thread ----
    const int rb = blockIdx.y * 128;
    const int cb = blockIdx.x * 64;
    const f4* tl = (const f4*)tile;
#pragma unroll
    for (int i = 0; i < 8; ++i) {
        const int idx = threadIdx.x + 256 * i;  // float4 index in [0, 2048)
        const int row = idx >> 4;               // 16 float4 per 64-col row
        const int c4  = idx & 15;
        f4 val = tl[idx];
        f4* dst = (f4*)(out + (size_t)(rb + row) * N_ROWS + cb + c4 * 4);
        __builtin_nontemporal_store(val, dst);
    }
}

// ---------------------------------------------------------------------------
// Launcher
// ---------------------------------------------------------------------------
extern "C" void kernel_launch(void* const* d_in, const int* in_sizes, int n_in,
                              void* d_out, int out_size, void* d_ws, size_t ws_size,
                              hipStream_t stream) {
    (void)in_sizes; (void)n_in; (void)out_size; (void)ws_size;

    const float* x1 = (const float*)d_in[0];
    const float* x2 = (const float*)d_in[1];
    float* out = (float*)d_out;

    // Workspace layout (bytes):
    //   [0)        x1hi  : 8192*128 bf16 = 2 MiB
    //   [2 MiB)    x1lo  : 2 MiB
    //   [4 MiB)    x2hi  : 2 MiB
    //   [6 MiB)    x2lo  : 2 MiB
    //   [8 MiB)    n1    : 8192 fp32
    //   [+32 KiB)  n2    : 8192 fp32
    const size_t MAT = (size_t)N_ROWS * K_DIM;           // elements per matrix
    char* ws = (char*)d_ws;
    __hip_bfloat16* x1hi = (__hip_bfloat16*)(ws);
    __hip_bfloat16* x1lo = (__hip_bfloat16*)(ws + 2 * MAT);
    __hip_bfloat16* x2hi = (__hip_bfloat16*)(ws + 4 * MAT);
    __hip_bfloat16* x2lo = (__hip_bfloat16*)(ws + 6 * MAT);
    float* n1 = (float*)(ws + 8 * MAT);
    float* n2 = n1 + N_ROWS;

    // Prep: one wave per row, 8 rows per block.
    dim3 pgrid(N_ROWS / 8), pblock(256);
    rbf_prep_kernel<<<pgrid, pblock, 0, stream>>>(x1, x1hi, x1lo, n1);
    rbf_prep_kernel<<<pgrid, pblock, 0, stream>>>(x2, x2hi, x2lo, n2);

    // GEMM: block tile 128 rows x 64 cols, 256 threads (8 waves).
    dim3 ggrid(N_ROWS / 64, N_ROWS / 128), gblock(256);
    rbf_gemm_kernel<<<ggrid, gblock, 0, stream>>>(x1hi, x1lo, x2hi, x2lo,
                                                  n1, n2, out);
}
